// Model_48361331753426
// MI455X (gfx1250) — matrix operand order repaired and checked by reading, not compiled, not run on hardware
//
#include <hip/hip_runtime.h>
#include <cstdint>

typedef _Float16 h16;
typedef __attribute__((ext_vector_type(16))) _Float16 v16h;
typedef __attribute__((ext_vector_type(8)))  _Float16 v8h;
typedef __attribute__((ext_vector_type(8)))  float    v8f;

#define WPB 4               // waves per block (128 threads)
#define PI_F 3.14159265358979f

#define N_BATCH   2048
#define WINDOW_N  16384
#define HOP0      512
#define REGIONS_N 33
#define FEAT_K    2772      // 84*33
#define FEAT_KP   2784      // padded to 87*32 (16B-aligned f16 row stride)
#define M_OUT_N   128
#define DS_STRIDE 16128     // 8192+4096+2048+1024+512+256 per batch
#define LPS_K     288       // 256-tap lowpass + shifts, rounded to 9*32
#define CQT_K     256

// ---- workspace byte offsets ----
#define OFF_LP    ((size_t)0)
#define OFF_LPS   ((size_t)1024)                          // 288*16 h16 = 9216B
#define OFF_CQTW  ((size_t)10240)                         // 256*32 h16 = 16384B
#define OFF_XH    ((size_t)32768)                         // 2048*16384 h16
#define OFF_DS    (OFF_XH + (size_t)N_BATCH*WINDOW_N*2)   // 2048*16128 h16
#define OFF_FEAT  (OFF_DS + (size_t)N_BATCH*DS_STRIDE*2)  // 2048*2784 h16
#define OFF_WH    (OFF_FEAT + (size_t)N_BATCH*FEAT_KP*2)  // 128*2784 h16

// Assemble an A fragment (two aligned 16B runs: [0..7], [8..15]) -> v16h
static __device__ __forceinline__ v16h frag16(const h16* p) {
  v8h lo = *(const v8h*)p;
  v8h hi = *(const v8h*)(p + 8);
  return __builtin_shufflevector(lo, hi, 0, 1, 2, 3, 4, 5, 6, 7,
                                         8, 9, 10, 11, 12, 13, 14, 15);
}

// ============================================================
// Init: synthesize lowpass + shifted-lowpass B matrix + CQT kernels
// ============================================================
__global__ void cqt_init_tables(float* lp, h16* lps, h16* cqtw) {
  __shared__ float s_h[256];
  __shared__ float s_sum;
  const int t = threadIdx.x;

  // Hamming-windowed sinc, band_center = 0.5
  float nn = (float)t - 127.5f;
  float xx = 0.5f * nn;
  float sc = sinf(PI_F * xx) / (PI_F * xx);
  float ham = 0.54f - 0.46f * cosf(2.0f * PI_F * (float)t / 255.0f);
  s_h[t] = 0.5f * sc * ham;
  __syncthreads();
  if (t == 0) { float s = 0.0f; for (int i = 0; i < 256; i++) s += s_h[i]; s_sum = s; }
  __syncthreads();
  float hn = s_h[t] / s_sum;
  lp[t] = hn;
  s_h[t] = hn;
  __syncthreads();

  // Shifted-lowpass matrix (A base = 2*m0-128): LPS[k][c] = LP[k-2c-1]
  for (int idx = t; idx < LPS_K * 16; idx += 256) {
    int k = idx >> 4, c = idx & 15, d = k - 2 * c - 1;
    lps[idx] = (d >= 0 && d < 256) ? (h16)s_h[d] : (h16)0.0f;
  }
  // zero-fill CQT weight matrix (cols 24..31 are padding)
  for (int idx = t; idx < CQT_K * 32; idx += 256) cqtw[idx] = (h16)0.0f;
  __syncthreads();

  // CQT kernels: 12 bins, Hann * complex exponential, L1-normalized (1/l cancels)
  if (t < 12) {
    float f = 3520.0f * exp2f((float)t / 12.0f);
    const float Qf = 16.81715374510576f;          // 1/(2^(1/12)-1)
    int l = (int)ceilf(Qf * 44100.0f / f);
    int start = (int)ceilf(128.0f - 0.5f * (float)l);
    float wsum = 0.0f;
    for (int n = 0; n < l; n++) wsum += 0.5f - 0.5f * cosf(2.0f * PI_F * (float)n / (float)l);
    float inv = 1.0f / wsum;
    for (int n = 0; n < l; n++) {
      float win = 0.5f - 0.5f * cosf(2.0f * PI_F * (float)n / (float)l);
      float th  = 2.0f * PI_F * (float)n * f / 44100.0f;
      int row = start + n;
      cqtw[row * 32 + t]      = (h16)(win * cosf(th) * inv);
      cqtw[row * 32 + 12 + t] = (h16)(win * sinf(th) * inv);
    }
  }
}

// ============================================================
// Small prep kernels
// ============================================================
__global__ void cqt_f32_to_f16(const float* __restrict__ x, h16* __restrict__ xh, int n) {
  int i = blockIdx.x * blockDim.x + threadIdx.x;
  if (i < n) xh[i] = (h16)x[i];
}

// W (f32 128x2772) -> Wh (f16 128x2784, zero-padded)
__global__ void cqt_prep_w(const float* __restrict__ W, h16* __restrict__ Wh) {
  int i = blockIdx.x * blockDim.x + threadIdx.x;
  if (i >= M_OUT_N * FEAT_KP) return;
  int o = i / FEAT_KP, k = i % FEAT_KP;
  Wh[i] = (k < FEAT_K) ? (h16)W[(size_t)o * FEAT_K + k] : (h16)0.0f;
}

// zero the 12 pad columns of each feat row
__global__ void cqt_zero_featpad(h16* __restrict__ feat) {
  int i = blockIdx.x * blockDim.x + threadIdx.x;
  if (i >= N_BATCH * (FEAT_KP - FEAT_K)) return;
  int b = i / (FEAT_KP - FEAT_K), k = FEAT_K + i % (FEAT_KP - FEAT_K);
  feat[(size_t)b * FEAT_KP + k] = (h16)0.0f;
}

// ============================================================
// Decimator stage as WMMA GEMM: y[m0+c] = sum_k x[2*m0-128+k] * LPS[k][c]
// Block = 4 consecutive 16x16 tiles of one batch; signal span staged in LDS.
// ============================================================
__global__ __launch_bounds__(WPB * 32) void cqt_downsample_wmma(
    const h16* __restrict__ src, int srcStride, int Lin,
    h16* __restrict__ dst, int dstStride,
    const h16* __restrict__ lps, int tilesPerBatch, int blocksPerBatch)
{
  __shared__ v16h sBf[(LPS_K / 32) * 32];   // 9 K-blocks * 32 lanes = 9216B
  __shared__ __align__(16) h16 sx[2304];    // 4*512 + 256 halves = 4608B

  const int batch = blockIdx.x / blocksPerBatch;
  const int t0    = (blockIdx.x % blocksPerBatch) * 4;
  const h16* sb = src + (size_t)batch * srcStride;
  const int gstart = 512 * t0 - 128;

  // B fragments into LDS (per-lane contiguous 32B)
  {
    h16* sB = (h16*)sBf;
    for (int i = threadIdx.x; i < LPS_K * 16; i += blockDim.x) {
      int blk = i >> 9, rem = i & 511, l = rem >> 4, j = rem & 15;
      int K = blk * 32 + (l >> 4) * 16 + j, c = l & 15;
      sB[i] = lps[K * 16 + c];
    }
  }
  // Signal span into LDS: 288 chunks of 8 halves (16B aligned both sides)
  for (int c = threadIdx.x; c < 2304 / 8; c += blockDim.x) {
    int g = gstart + c * 8;
    if (g >= 0 && g + 8 <= Lin) {
      *(v8h*)(sx + c * 8) = *(const v8h*)(sb + g);
    } else {
#pragma unroll
      for (int e = 0; e < 8; e++) {
        int gi = g + e;
        sx[c * 8 + e] = (gi >= 0 && gi < Lin) ? sb[gi] : (h16)0.0f;
      }
    }
  }
  __syncthreads();

  const int wave = threadIdx.x >> 5, lane = threadIdx.x & 31;
  if (t0 + wave >= tilesPerBatch) return;
  const int trow = t0 + wave;
  const int low = lane & 15, hi = lane >> 4;
  const int aOff = 512 * wave + 32 * low + hi * 16;   // 16B aligned

  v8f acc = {};
#pragma unroll
  for (int kk = 0; kk < LPS_K; kk += 32) {
    v16h a = frag16(sx + aOff + kk);
    v16h b = sBf[(kk >> 5) * 32 + lane];
    acc = __builtin_amdgcn_wmma_f32_16x16x32_f16(false, a, false, b, (short)0, acc,
                                                 false, false);
  }

  h16* db = dst + (size_t)batch * dstStride;
#pragma unroll
  for (int rr = 0; rr < 8; rr++) {
    int m = (trow * 16 + rr + hi * 8) * 16 + low;
    db[m] = (h16)acc[rr];
  }
}

// ============================================================
// One CQT octave: block = one batch; padded signal staged in dynamic LDS.
// 3 waves cover the 33 time rows (rows >=33 clamped, not stored).
// GEMM (48 x 256 x 24pad32) -> magnitude -> log -> feat (f16).
// ============================================================
__global__ __launch_bounds__(WPB * 32) void cqt_octave_wmma(
    const h16* __restrict__ sig, int sigStride, int L, int hop,
    const h16* __restrict__ cqtw, h16* __restrict__ feat, int binBase)
{
  extern __shared__ __align__(16) char smem[];
  v16h* sWf = (v16h*)smem;                               // 512 v16h = 16384B
  float* sDall = (float*)(smem + 16384);                 // WPB*16*24 f32 = 6144B
  h16*  sx  = (h16*)(smem + 16384 + WPB * 16 * 24 * 4);  // L+256 halves

  const int b = blockIdx.x;
  const h16* sb = sig + (size_t)b * sigStride;

  // B fragments (2 col-tiles) into LDS
  {
    h16* sW = (h16*)sWf;
    for (int i = threadIdx.x; i < CQT_K * 32; i += blockDim.x) {
      int blk = i >> 10, tle = (i >> 9) & 1, l = (i >> 4) & 31, j = i & 15;
      int K = blk * 32 + (l >> 4) * 16 + j;
      int col = tle * 16 + (l & 15);
      sW[i] = cqtw[K * 32 + col];
    }
  }
  // Padded signal into LDS: element g = c*8 - 128
  const int nChunks = (L + 256) / 8;
  for (int c = threadIdx.x; c < nChunks; c += blockDim.x) {
    int g = c * 8 - 128;
    if (g >= 0 && g + 8 <= L) {
      *(v8h*)(sx + c * 8) = *(const v8h*)(sb + g);
    } else {
#pragma unroll
      for (int e = 0; e < 8; e++) {
        int gi = g + e;
        sx[c * 8 + e] = (gi >= 0 && gi < L) ? sb[gi] : (h16)0.0f;
      }
    }
  }
  __syncthreads();

  const int wave = threadIdx.x >> 5, lane = threadIdx.x & 31;
  if (wave >= 3) return;
  const int t0 = wave * 16;
  const int low = lane & 15, hi = lane >> 4;
  const int tt = (t0 + low < REGIONS_N) ? (t0 + low) : (REGIONS_N - 1);
  const int aOff = tt * hop + hi * 16;                    // 16B aligned (hop >= 8)

  v8f acc0 = {}, acc1 = {};
#pragma unroll
  for (int kk = 0; kk < CQT_K; kk += 32) {
    v16h a  = frag16(sx + aOff + kk);
    int blk = kk >> 5;
    v16h b0 = sWf[blk * 64 + lane];
    v16h b1 = sWf[blk * 64 + 32 + lane];
    acc0 = __builtin_amdgcn_wmma_f32_16x16x32_f16(false, a, false, b0, (short)0, acc0,
                                                  false, false);
    acc1 = __builtin_amdgcn_wmma_f32_16x16x32_f16(false, a, false, b1, (short)0, acc1,
                                                  false, false);
  }

  // Stash D tile (16 rows x 24 used cols) into wave-private LDS
  float* dt = sDall + wave * (16 * 24);
#pragma unroll
  for (int rr = 0; rr < 8; rr++) {
    int row = rr + hi * 8;
    dt[row * 24 + low] = acc0[rr];                      // cols 0..15
    if (low < 8) dt[row * 24 + 16 + low] = acc1[rr];    // cols 16..23
  }
  // 16 rows * 12 bins; LDS ops are in-order within a wave
  for (int idx = lane; idx < 192; idx += 32) {
    int row = idx / 12, k = idx % 12;
    int t = t0 + row;
    if (t >= REGIONS_N) continue;
    float re = dt[row * 24 + k];
    float im = dt[row * 24 + 12 + k];
    float mag = sqrtf(re * re + im * im);
    float fv  = logf(mag + 1e-5f);
    feat[(size_t)b * FEAT_KP + (size_t)(binBase + k) * REGIONS_N + t] = (h16)fv;
  }
}

// ============================================================
// Head: out = sigmoid(feat @ Wh^T), 2048 x 128 x 2784 (both f16, zero-padded)
// Guard-free uniform K loop, all operand loads are aligned b128s.
// ============================================================
__global__ __launch_bounds__(WPB * 32) void cqt_linear_sigmoid_wmma(
    const h16* __restrict__ feat, const h16* __restrict__ Wh,
    float* __restrict__ out)
{
  const int wave = threadIdx.x >> 5, lane = threadIdx.x & 31;
  int tile = blockIdx.x * WPB + wave;
  if (tile >= (N_BATCH / 16) * (M_OUT_N / 16)) return;
  int mt = tile >> 3, nt = tile & 7;

  const int low = lane & 15, hi = lane >> 4;
  const h16* fb = feat + (size_t)(mt * 16 + low) * FEAT_KP + hi * 16;
  const h16* wb = Wh   + (size_t)(nt * 16 + low) * FEAT_KP + hi * 16;

  v8f acc = {};
  for (int kk = 0; kk < FEAT_KP; kk += 32) {
    v16h a = frag16(fb + kk);              // rows of feat (A)
    v16h bm = *(const v16h*)(wb + kk);     // 16 contiguous K for this out-col (B)
    acc = __builtin_amdgcn_wmma_f32_16x16x32_f16(false, a, false, bm, (short)0, acc,
                                                 false, false);
  }
#pragma unroll
  for (int rr = 0; rr < 8; rr++) {
    int row = mt * 16 + rr + hi * 8;
    float v = acc[rr];
    out[(size_t)row * M_OUT_N + nt * 16 + low] = 1.0f / (1.0f + expf(-v));
  }
}

// ============================================================
extern "C" void kernel_launch(void* const* d_in, const int* in_sizes, int n_in,
                              void* d_out, int out_size, void* d_ws, size_t ws_size,
                              hipStream_t stream) {
  (void)in_sizes; (void)n_in; (void)out_size; (void)ws_size;
  const float* x = (const float*)d_in[0];
  const float* W = (const float*)d_in[1];
  float* out = (float*)d_out;

  char* ws   = (char*)d_ws;
  float* lp  = (float*)(ws + OFF_LP);
  h16* lps   = (h16*)(ws + OFF_LPS);
  h16* cqtw  = (h16*)(ws + OFF_CQTW);
  h16* xh    = (h16*)(ws + OFF_XH);
  h16* dsbuf = (h16*)(ws + OFF_DS);
  h16* feat  = (h16*)(ws + OFF_FEAT);
  h16* wh    = (h16*)(ws + OFF_WH);

  cqt_init_tables<<<1, 256, 0, stream>>>(lp, lps, cqtw);
  {
    int n = N_BATCH * WINDOW_N;
    cqt_f32_to_f16<<<n / 256, 256, 0, stream>>>(x, xh, n);
  }
  cqt_prep_w<<<(M_OUT_N * FEAT_KP) / 256, 256, 0, stream>>>(W, wh);
  cqt_zero_featpad<<<(N_BATCH * (FEAT_KP - FEAT_K)) / 256, 256, 0, stream>>>(feat);

  // per-batch offsets of each octave's decimated signal inside DS region
  static const int dsOff[7] = {0, 0, 8192, 12288, 14336, 15360, 15872};

  // decimation chain (octave 1..6)
  for (int oct = 1; oct <= 6; oct++) {
    int Lin = WINDOW_N >> (oct - 1);
    int Lo  = Lin >> 1;
    const h16* src; int sStride;
    if (oct == 1) { src = xh; sStride = WINDOW_N; }
    else          { src = dsbuf + dsOff[oct - 1]; sStride = DS_STRIDE; }
    h16* dst = dsbuf + dsOff[oct];
    int tilesPerBatch  = Lo / 256;                    // 16 rows * 16 outputs per tile
    int blocksPerBatch = (tilesPerBatch + 3) / 4;
    cqt_downsample_wmma<<<N_BATCH * blocksPerBatch, WPB * 32, 0, stream>>>(
        src, sStride, Lin, dst, DS_STRIDE, lps, tilesPerBatch, blocksPerBatch);
  }

  // CQT per octave (feat rows: octave 6 -> bins 0..11, ..., octave 0 -> 72..83)
  for (int oct = 0; oct < 7; oct++) {
    const h16* sig; int sStride, L;
    if (oct == 0) { sig = xh; sStride = WINDOW_N; L = WINDOW_N; }
    else          { sig = dsbuf + dsOff[oct]; sStride = DS_STRIDE; L = WINDOW_N >> oct; }
    int hop = HOP0 >> oct;
    size_t smemBytes = 16384 + (size_t)WPB * 16 * 24 * 4 + (size_t)(L + 256) * 2;
    cqt_octave_wmma<<<N_BATCH, WPB * 32, smemBytes, stream>>>(
        sig, sStride, L, hop, cqtw, feat, (6 - oct) * 12);
  }

  // final linear + sigmoid
  {
    int tiles = (N_BATCH / 16) * (M_OUT_N / 16);  // 1024
    cqt_linear_sigmoid_wmma<<<tiles / WPB, WPB * 32, 0, stream>>>(feat, wh, out);
  }
}
